// MinGRULM_33260226740633
// MI455X (gfx1250) — compile-verified, matrix operand order checked by reading, not run on hardware
//
#include <hip/hip_runtime.h>
#include <hip/hip_bf16.h>
#include <math.h>

// ---------------------------------------------------------------------------
// MinGRU LM forward for MI455X (gfx1250), wave32 + WMMA bf16.
//
// Roofline reasoning (cannot measure; from specs):
//  * 263 GFLOP total, dominated by GEMMs (K=512/2048, N up to 32000).
//  * HBM 23.3 TB/s: the 524 MB logits write costs ~22us; weights (~84MB)
//    fit in 192MB L2, so GEMMs are matrix-pipe bound -> use
//    v_wmma_f32_16x16x32_bf16 (bf16 operands, f32 accum).
//  * minGRU scan = associative logaddexp linear recurrence -> 3-phase
//    chunked parallel scan (32 chunks x 128 steps x 512 channels).
//  * A-tile staging uses gfx1250 async global->LDS copies (ASYNCcnt) when
//    the toolchain exposes the builtin; falls back to manual staging.
// ---------------------------------------------------------------------------

typedef __attribute__((ext_vector_type(16))) __bf16 v16bf;
typedef __attribute__((ext_vector_type(8)))  float  v8f;
typedef __attribute__((ext_vector_type(4)))  int    v4i;

#define D_MODEL   512
#define SEQ       4096
#define NLAYER    6
#define VOCAB     32000
#define FFDIM     2048      // MULT * D
#define SCAN_CHUNKS 32
#define SCAN_LEN    128     // SEQ / SCAN_CHUNKS

#if __has_builtin(__builtin_amdgcn_global_load_async_to_lds_b128) && \
    __has_builtin(__builtin_amdgcn_s_wait_asynccnt)
#define USE_ASYNC_STAGE 1
#else
#define USE_ASYNC_STAGE 0
#endif

// ------------------------------ math helpers -------------------------------
__device__ __forceinline__ float softplus_f(float x) {
    // stable: max(x,0) + log1p(exp(-|x|))
    return fmaxf(x, 0.0f) + log1pf(expf(-fabsf(x)));
}
__device__ __forceinline__ float log_g_f(float x) {
    return (x >= 0.0f) ? logf(fmaxf(x, 0.0f) + 0.5f) : -softplus_f(-x);
}
__device__ __forceinline__ float logaddexp_f(float a, float b) {
    float m = fmaxf(a, b);
    if (m == -INFINITY) return -INFINITY;
    return m + log1pf(expf(fminf(a, b) - m));
}

// ------------------------------- embedding ---------------------------------
__global__ __launch_bounds__(256) void embed_kernel(
    const int* __restrict__ ids, const float* __restrict__ emb,
    float* __restrict__ h)
{
    size_t i = (size_t)blockIdx.x * 256 + threadIdx.x;   // over SEQ*D
    int s = (int)(i / D_MODEL);
    int d = (int)(i % D_MODEL);
    h[i] = emb[(size_t)ids[s] * D_MODEL + d];
}

// --------------------------- rmsnorm -> bf16 -------------------------------
// y = x / ||x|| * sqrt(D) * (g+1)  ==  x * rsqrt(sumsq/D) * (g+1)
__global__ __launch_bounds__(256) void rmsnorm_bf16_kernel(
    const float* __restrict__ x, const float* __restrict__ g,
    __bf16* __restrict__ y)
{
    const int lane = threadIdx.x & 31;
    const int wave = threadIdx.x >> 5;
    const int tok  = blockIdx.x * 8 + wave;
    const float* xr = x + (size_t)tok * D_MODEL;

    float xs[16];
    float ss = 0.0f;
#pragma unroll
    for (int i = 0; i < 16; ++i) {
        float v = xr[i * 32 + lane];
        xs[i] = v;
        ss += v * v;
    }
#pragma unroll
    for (int off = 16; off > 0; off >>= 1) ss += __shfl_xor(ss, off, 32);

    const float inv = rsqrtf(fmaxf(ss, 1e-24f) / (float)D_MODEL);
    __bf16* yr = y + (size_t)tok * D_MODEL;
#pragma unroll
    for (int i = 0; i < 16; ++i) {
        int c = i * 32 + lane;
        yr[c] = (__bf16)(xs[i] * inv * (g[c] + 1.0f));
    }
}

// --------------------------- f32 -> bf16 convert ---------------------------
__global__ __launch_bounds__(256) void cvt_bf16_kernel(
    const float* __restrict__ src, __bf16* __restrict__ dst, size_t n)
{
    size_t i = (size_t)blockIdx.x * 256 + threadIdx.x;
    if (i < n) dst[i] = (__bf16)src[i];
}

// ------------------------------ WMMA GEMM ----------------------------------
// C[M,N] = A[M,K](bf16, row-major) * B[K,N](bf16, row-major), f32 out.
// Block tile 128x128x32, 8 waves; wave tile 32x64 = 2x4 WMMA 16x16 tiles.
// M,N multiples of 128; K multiple of 32 (true for all shapes here).
__global__ __launch_bounds__(256) void gemm_bf16_wmma_kernel(
    const __bf16* __restrict__ A, const __bf16* __restrict__ B,
    float* __restrict__ C, int M, int N, int K)
{
    __shared__ __bf16 As[128][32];    // [m][k]
    __shared__ __bf16 BsT[128][32];   // [n][k]  (transposed stage)

    const int tid  = threadIdx.x;
    const int lane = tid & 31;
    const int wave = tid >> 5;
    const int wm = wave & 3;          // 4 waves along M
    const int wn = wave >> 2;         // 2 waves along N
    const int bM = blockIdx.y * 128;
    const int bN = blockIdx.x * 128;

    v8f acc[2][4];
#pragma unroll
    for (int im = 0; im < 2; ++im)
#pragma unroll
        for (int jn = 0; jn < 4; ++jn) acc[im][jn] = (v8f){0,0,0,0,0,0,0,0};

    const int aRow   = tid >> 1;      // 0..127
    const int aHalf  = tid & 1;       // 16 bf16 per half-row
    const int bRow   = tid >> 3;      // 0..31 (k)
    const int bChunk = tid & 7;       // 8 x 16 cols

    for (int k0 = 0; k0 < K; k0 += 32) {
        // ---- stage A tile: 128 rows x 32 k, 16 bf16 (32B) per thread ----
#if USE_ASYNC_STAGE
        {
            // gfx1250 async direct global->LDS copy, tracked by ASYNCcnt
            const __bf16* gsrc = A + (size_t)(bM + aRow) * K + k0 + aHalf * 16;
            __builtin_amdgcn_global_load_async_to_lds_b128(
                (__attribute__((address_space(1))) v4i*)gsrc,
                (__attribute__((address_space(3))) v4i*)&As[aRow][aHalf * 16],
                0, 0);
            __builtin_amdgcn_global_load_async_to_lds_b128(
                (__attribute__((address_space(1))) v4i*)(gsrc + 8),
                (__attribute__((address_space(3))) v4i*)&As[aRow][aHalf * 16 + 8],
                0, 0);
        }
#else
        {
            uint4 av = *(const uint4*)(A + (size_t)(bM + aRow) * K + k0 + aHalf * 16);
            *(uint4*)(&As[aRow][aHalf * 16]) = av;
        }
#endif

        // ---- stage B tile transposed: read 16 consecutive n for one k row ----
        uint4 bv = *(const uint4*)(B + (size_t)(k0 + bRow) * N + bN + bChunk * 16);
        const __bf16* bvp = (const __bf16*)&bv;
#pragma unroll
        for (int i = 0; i < 16; ++i) BsT[bChunk * 16 + i][bRow] = bvp[i];

#if USE_ASYNC_STAGE
        __builtin_amdgcn_s_wait_asynccnt(0);
#endif
        __syncthreads();

        // hint-prefetch next A slab (global_prefetch_b8 on gfx1250)
        if (k0 + 32 < K)
            __builtin_prefetch(A + (size_t)(bM + aRow) * K + k0 + 32, 0, 0);

        // load A fragments (ISA 16-bit A layout: K pairs per dword;
        // lanes 0-15 hold K {0..7,16..23}, lanes 16-31 hold {8..15,24..31})
        v16bf afrag[2];
#pragma unroll
        for (int im = 0; im < 2; ++im) {
            const int m = wm * 32 + im * 16 + (lane & 15);
#pragma unroll
            for (int vg = 0; vg < 8; ++vg) {
                const int kk = ((vg & 4) << 2) + ((lane >> 4) << 3) + ((vg & 3) << 1);
                afrag[im][2 * vg]     = As[m][kk];
                afrag[im][2 * vg + 1] = As[m][kk + 1];
            }
        }

#pragma unroll
        for (int jn = 0; jn < 4; ++jn) {
            v16bf bfrag;
            const int n = wn * 64 + jn * 16 + (lane & 15);
#pragma unroll
            for (int vg = 0; vg < 8; ++vg) {
                const int kk = ((vg & 4) << 2) + ((lane >> 4) << 3) + ((vg & 3) << 1);
                bfrag[2 * vg]     = BsT[n][kk];
                bfrag[2 * vg + 1] = BsT[n][kk + 1];
            }
#pragma unroll
            for (int im = 0; im < 2; ++im) {
                acc[im][jn] = __builtin_amdgcn_wmma_f32_16x16x32_bf16(
                    false, afrag[im], false, bfrag,
                    (short)0, acc[im][jn], false, false);
            }
        }
        __syncthreads();
    }

    // epilogue: VGPR r -> (M=r, N=lane) lanes 0-15 ; (M=r+8, N=lane-16)
    const int mAdd = (lane >> 4) * 8;
    const int nCol = lane & 15;
#pragma unroll
    for (int im = 0; im < 2; ++im)
#pragma unroll
        for (int jn = 0; jn < 4; ++jn)
#pragma unroll
            for (int r = 0; r < 8; ++r) {
                const int m = bM + wm * 32 + im * 16 + r + mAdd;
                const int n = bN + wn * 64 + jn * 16 + nCol;
                C[(size_t)m * N + n] = acc[im][jn][r];
            }
}

// ----------------------- minGRU scan (3-phase, log-space) ------------------
// recurrence: log_h_t = logaddexp(lc_t + log_h_{t-1}, lv_t)
// operator (A1,B1) o (A2,B2) = (A1+A2, logaddexp(B1+A2, B2))  [associative]
__global__ __launch_bounds__(256) void scan_phase1_kernel(
    const float* __restrict__ hg,   // [SEQ, 2D]: hidden | gate
    float* __restrict__ Al, float* __restrict__ Bl,
    float* __restrict__ At, float* __restrict__ Bt)
{
    const int ch    = blockIdx.y * 256 + threadIdx.x;  // 0..511
    const int chunk = blockIdx.x;
    float A  = 0.0f;
    float Bv = -INFINITY;
    for (int t = 0; t < SCAN_LEN; ++t) {
        const int s = chunk * SCAN_LEN + t;
        const float hid  = hg[(size_t)s * (2 * D_MODEL) + ch];
        const float gate = hg[(size_t)s * (2 * D_MODEL) + D_MODEL + ch];
        const float lc = -softplus_f(gate);                  // log(1-z)
        const float lv = -softplus_f(-gate) + log_g_f(hid);  // log z + log g
        A  += lc;
        Bv  = logaddexp_f(Bv + lc, lv);
        Al[(size_t)s * D_MODEL + ch] = A;
        Bl[(size_t)s * D_MODEL + ch] = Bv;
    }
    At[chunk * D_MODEL + ch] = A;
    Bt[chunk * D_MODEL + ch] = Bv;
}

__global__ __launch_bounds__(512) void scan_phase2_kernel(
    const float* __restrict__ At, const float* __restrict__ Bt,
    float* __restrict__ Pb)
{
    const int ch = threadIdx.x;  // 0..511
    float Bv = -INFINITY;
    for (int c = 0; c < SCAN_CHUNKS; ++c) {
        Pb[c * D_MODEL + ch] = Bv;                 // exclusive prefix
        Bv = logaddexp_f(Bv + At[c * D_MODEL + ch], Bt[c * D_MODEL + ch]);
    }
}

__global__ __launch_bounds__(256) void scan_phase3_kernel(
    const float* __restrict__ Al, const float* __restrict__ Bl,
    const float* __restrict__ Pb, float* __restrict__ h)
{
    size_t i = (size_t)blockIdx.x * 256 + threadIdx.x;   // over SEQ*D
    const int s     = (int)(i / D_MODEL);
    const int chunk = s / SCAN_LEN;
    const int ch    = (int)(i % D_MODEL);
    const float lh = logaddexp_f(Pb[chunk * D_MODEL + ch] + Al[i], Bl[i]);
    h[i] += expf(lh);    // residual: minGRU(x) + h
}

// ---------------------------- FF epilogues ---------------------------------
__global__ __launch_bounds__(256) void bias_gelu_bf16_kernel(
    const float* __restrict__ C, const float* __restrict__ b,
    __bf16* __restrict__ out, int ncols, size_t n)
{
    size_t i = (size_t)blockIdx.x * 256 + threadIdx.x;
    if (i >= n) return;
    const float v = C[i] + b[i % ncols];
    out[i] = (__bf16)(0.5f * v * (1.0f + erff(v * 0.70710678118654752f)));
}

__global__ __launch_bounds__(256) void residual_bias_kernel(
    float* __restrict__ h, const float* __restrict__ C,
    const float* __restrict__ b)
{
    size_t i = (size_t)blockIdx.x * 256 + threadIdx.x;   // over SEQ*D
    h[i] += C[i] + b[i % D_MODEL];
}

// ------------------------------- launcher ----------------------------------
extern "C" void kernel_launch(void* const* d_in, const int* in_sizes, int n_in,
                              void* d_out, int out_size, void* d_ws, size_t ws_size,
                              hipStream_t stream) {
    const int*   ids     = (const int*)  d_in[0];
    const float* emb     = (const float*)d_in[1];
    const float* norm1_g = (const float*)d_in[2];
    const float* w_hg    = (const float*)d_in[3];
    const float* norm2_g = (const float*)d_in[4];
    const float* ff_w1   = (const float*)d_in[5];
    const float* ff_b1   = (const float*)d_in[6];
    const float* ff_w2   = (const float*)d_in[7];
    const float* ff_b2   = (const float*)d_in[8];
    const float* final_g = (const float*)d_in[9];
    const float* out_w   = (const float*)d_in[10];
    float* logits = (float*)d_out;

    // workspace layout (all offsets 256B-aligned), ~104 MiB total
    char* p = (char*)d_ws;
    float*  h    = (float*) p;  p += (size_t)SEQ * D_MODEL * 4;        // 8 MB
    __bf16* abuf = (__bf16*)p;  p += (size_t)SEQ * FFDIM * 2;          // 16 MB (A operand / act)
    __bf16* wb   = (__bf16*)p;  p += (size_t)D_MODEL * VOCAB * 2;      // 32 MB (bf16 weights)
    float*  Cbuf = (float*) p;  p += (size_t)SEQ * FFDIM * 4;          // 33.5 MB (GEMM out)
    float*  Al   = (float*) p;  p += (size_t)SEQ * D_MODEL * 4;        // 8 MB
    float*  Bl   = (float*) p;  p += (size_t)SEQ * D_MODEL * 4;        // 8 MB
    float*  At   = (float*) p;  p += (size_t)SCAN_CHUNKS * D_MODEL * 4;
    float*  Bt   = (float*) p;  p += (size_t)SCAN_CHUNKS * D_MODEL * 4;
    float*  Pb   = (float*) p;  p += (size_t)SCAN_CHUNKS * D_MODEL * 4;

    const dim3 blk(256);
    const int  nSD   = (SEQ * D_MODEL) / 256;     // elementwise over [S,D]
    const int  nNorm = SEQ / 8;                   // 8 tokens per block

    // h = emb[x]
    embed_kernel<<<nSD, blk, 0, stream>>>(ids, emb, h);

    for (int l = 0; l < NLAYER; ++l) {
        // ---- minGRU branch ----
        rmsnorm_bf16_kernel<<<nNorm, blk, 0, stream>>>(h, norm1_g + (size_t)l * D_MODEL, abuf);
        {
            size_t n = (size_t)D_MODEL * 2 * D_MODEL;
            cvt_bf16_kernel<<<(unsigned)((n + 255) / 256), blk, 0, stream>>>(
                w_hg + (size_t)l * n, wb, n);
        }
        gemm_bf16_wmma_kernel<<<dim3((2 * D_MODEL) / 128, SEQ / 128), blk, 0, stream>>>(
            abuf, wb, Cbuf, SEQ, 2 * D_MODEL, D_MODEL);

        scan_phase1_kernel<<<dim3(SCAN_CHUNKS, D_MODEL / 256), blk, 0, stream>>>(
            Cbuf, Al, Bl, At, Bt);
        scan_phase2_kernel<<<1, 512, 0, stream>>>(At, Bt, Pb);
        scan_phase3_kernel<<<nSD, blk, 0, stream>>>(Al, Bl, Pb, h);

        // ---- FF branch ----
        rmsnorm_bf16_kernel<<<nNorm, blk, 0, stream>>>(h, norm2_g + (size_t)l * D_MODEL, abuf);
        {
            size_t n = (size_t)D_MODEL * FFDIM;
            cvt_bf16_kernel<<<(unsigned)((n + 255) / 256), blk, 0, stream>>>(
                ff_w1 + (size_t)l * n, wb, n);
        }
        gemm_bf16_wmma_kernel<<<dim3(FFDIM / 128, SEQ / 128), blk, 0, stream>>>(
            abuf, wb, Cbuf, SEQ, FFDIM, D_MODEL);

        bias_gelu_bf16_kernel<<<(SEQ * FFDIM) / 256, blk, 0, stream>>>(
            Cbuf, ff_b1 + (size_t)l * FFDIM, abuf, FFDIM, (size_t)SEQ * FFDIM);

        {
            size_t n = (size_t)FFDIM * D_MODEL;
            cvt_bf16_kernel<<<(unsigned)((n + 255) / 256), blk, 0, stream>>>(
                ff_w2 + (size_t)l * n, wb, n);
        }
        gemm_bf16_wmma_kernel<<<dim3(D_MODEL / 128, SEQ / 128), blk, 0, stream>>>(
            abuf, wb, Cbuf, SEQ, D_MODEL, FFDIM);

        residual_bias_kernel<<<nSD, blk, 0, stream>>>(h, Cbuf, ff_b2 + (size_t)l * D_MODEL);
    }

    // ---- final norm + logits ----
    rmsnorm_bf16_kernel<<<nNorm, blk, 0, stream>>>(h, final_g, abuf);
    {
        size_t n = (size_t)D_MODEL * VOCAB;
        cvt_bf16_kernel<<<(unsigned)((n + 255) / 256), blk, 0, stream>>>(out_w, wb, n);
    }
    gemm_bf16_wmma_kernel<<<dim3(VOCAB / 128, SEQ / 128), blk, 0, stream>>>(
        abuf, wb, logits, SEQ, VOCAB, D_MODEL);
}
